// RNNRBM_64613488001653
// MI455X (gfx1250) — matrix operand order repaired and checked by reading, not compiled, not run on hardware
//
#include <hip/hip_runtime.h>
#include <hip/hip_bf16.h>
#include <math.h>

// ---------------------------------------------------------------------------
// RNN-RBM for MI455X (gfx1250, wave32, WMMA bf16 + TDM async tensor loads)
// ---------------------------------------------------------------------------

typedef __bf16 bf16_t;
typedef __attribute__((ext_vector_type(16))) __bf16 v16bf;
typedef __attribute__((ext_vector_type(8)))  __bf16 v8bf;
typedef __attribute__((ext_vector_type(8)))  float  v8f;

#define T_ROWS 16384
#define VD 88
#define VDP 96
#define HD 512
#define RD 512
#define GIBBS 20
#define EPSV 1e-6f
#define GW 6       // waves per gibbs block
#define ISPLIT 304 // wuu columns resident in LDS during scan (multiple of 8)

// ---- workspace layout (bytes) ----
#define OFF_CIN   ((size_t)0)                          // 16384*512 f32
#define OFF_UBF   (OFF_CIN  + (size_t)16384*512*4)     // 16384*512 bf16
#define OFF_BH    (OFF_UBF  + (size_t)16384*512*2)     // 16384*512 f32
#define OFF_BV    (OFF_BH   + (size_t)16384*512*4)     // 16384*96  f32
#define OFF_WUUT  (OFF_BV   + (size_t)16384*96*4)      // 512*512 bf16 (col-major of wuu)
#define OFF_WUHT  (OFF_WUUT + (size_t)512*512*2)       // 512*512 bf16 (N-major)
#define OFF_WUVT  (OFF_WUHT + (size_t)512*512*2)       // 96*512  bf16 (N-major, padded)
#define OFF_WT    (OFF_WUVT + (size_t)96*512*2)        // 512*96  bf16 (w, HD-major)
#define OFF_WV    (OFF_WT   + (size_t)512*96*2)        // 96*512  bf16 (w, VD-major)

// ---------------------------------------------------------------------------
// Tensor Data Mover (TDM) support
// ---------------------------------------------------------------------------
#if defined(__has_builtin)
#  if __has_builtin(__builtin_amdgcn_tensor_load_to_lds)
#    define HAVE_TDM 1
#  endif
#endif
#ifndef HAVE_TDM
#  define HAVE_TDM 0
#endif

#if HAVE_TDM
typedef __attribute__((ext_vector_type(4))) unsigned v4u;
typedef __attribute__((ext_vector_type(4))) int      v4i_;
typedef __attribute__((ext_vector_type(8))) int      v8i_;

// 2D tile load: tile_d0 x tile_d1 elements (2 bytes each) out of a tensor with
// row stride stride0_elems; rows packed contiguously into LDS at lds_off.
// D# layout per CDNA5 ISA 8.3/8.4 (group0: count/lds/global/type, group1: dims).
__device__ __forceinline__ void tdm_load_2d(unsigned lds_off, const void* gptr,
                                            unsigned tile_d0, unsigned tile_d1,
                                            unsigned tensor_d0, unsigned tensor_d1,
                                            unsigned stride0_elems) {
  unsigned long long ga = (unsigned long long)(size_t)gptr;
  v4u g0;
  g0[0] = 1u;                                            // count=1 (valid), user mode
  g0[1] = lds_off;                                       // lds_addr (bytes)
  g0[2] = (unsigned)ga;                                  // global_addr[31:0]
  g0[3] = (unsigned)((ga >> 32) & 0x01FFFFFFu)           // global_addr[56:32]
        | (2u << 30);                                    // type = 2 ("image")
  v8i_ g1;
  g1[0] = (int)(1u << 16);                               // data_size=1 -> 2 bytes
  g1[1] = (int)((tensor_d0 & 0xFFFFu) << 16);            // tensor_dim0[15:0] @ 63:48
  g1[2] = (int)((tensor_d0 >> 16) |
                ((tensor_d1 & 0xFFFFu) << 16));          // dim0 hi | dim1 lo
  g1[3] = (int)(((tensor_d1 >> 16) & 0xFFFFu) |
                ((tile_d0 & 0xFFFFu) << 16));            // dim1 hi | tile_dim0
  g1[4] = (int)(tile_d1 & 0xFFFFu);                      // tile_dim1 (tile_dim2 = 0)
  g1[5] = (int)stride0_elems;                            // tensor_dim0_stride lo
  g1[6] = 0;                                             // stride0 hi | stride1 lo
  g1[7] = 0;
  v4i_ z4 = (v4i_)0;
#if defined(__clang_major__) && (__clang_major__ >= 23)
  v8i_ z8 = (v8i_)0;
  __builtin_amdgcn_tensor_load_to_lds(g0, g1, z4, z4, z8, 0);
#else
  __builtin_amdgcn_tensor_load_to_lds(g0, g1, z4, z4, 0);
#endif
}
#endif // HAVE_TDM

// ---------------------------------------------------------------------------
// WMMA helpers
// ---------------------------------------------------------------------------
__device__ __forceinline__ v8f wmma_bf16(v16bf a, v16bf b, v8f c) {
  return __builtin_amdgcn_wmma_f32_16x16x32_bf16(false, a, false, b, (short)0, c,
                                                 false, false);
}

// A-matrix 16x32 bf16 fragment: lane m=L&15, g=L>>4; K = g*8+{0..7}, 16+g*8+{0..7}
__device__ __forceinline__ v16bf load_fragA(const bf16_t* base, int row0, int ld,
                                            int k0, int lane) {
  const int m = lane & 15, g = lane >> 4;
  const bf16_t* p = base + (size_t)(row0 + m) * ld + k0 + g * 8;
  v8bf lo = *reinterpret_cast<const v8bf*>(p);
  v8bf hi = *reinterpret_cast<const v8bf*>(p + 16);
  return __builtin_shufflevector(lo, hi, 0,1,2,3,4,5,6,7,8,9,10,11,12,13,14,15);
}

// B-matrix 32x16 bf16 fragment from N-major (transposed) storage:
// lane n=L&15, g=L>>4; K = g*8+{0..7}, 16+g*8+{0..7}
__device__ __forceinline__ v16bf load_fragB(const bf16_t* baseT, int n0, int ld,
                                            int k0, int lane) {
  const int n = lane & 15, g = lane >> 4;
  const bf16_t* p = baseT + (size_t)(n0 + n) * ld + k0 + g * 8;
  v8bf lo = *reinterpret_cast<const v8bf*>(p);
  v8bf hi = *reinterpret_cast<const v8bf*>(p + 16);
  return __builtin_shufflevector(lo, hi, 0,1,2,3,4,5,6,7,8,9,10,11,12,13,14,15);
}

__device__ __forceinline__ float urand(unsigned r, unsigned c, unsigned s, unsigned ph) {
  unsigned x = r * 0x9E3779B1u ^ (c + 0x85EBCA77u) * 0xC2B2AE3Du
             ^ (((s << 1) | ph) + 0x27D4EB2Fu) * 0x165667B1u;
  x ^= x >> 16; x *= 0x7FEB352Du; x ^= x >> 15; x *= 0x846CA68Bu; x ^= x >> 16;
  return (float)(x >> 8) * (1.0f / 16777216.0f);
}

__device__ __forceinline__ float sigmoidf(float x) { return 1.0f / (1.0f + expf(-x)); }

// ---------------------------------------------------------------------------
// prep: bf16 transposed/padded weight copies + u pad rows
// ---------------------------------------------------------------------------
__global__ void prep_kernel(const float* __restrict__ w,   const float* __restrict__ wuu,
                            const float* __restrict__ wuv, const float* __restrict__ wuh,
                            const float* __restrict__ u0,
                            bf16_t* __restrict__ wuuT, bf16_t* __restrict__ wuhT,
                            bf16_t* __restrict__ wuvT, bf16_t* __restrict__ wt,
                            bf16_t* __restrict__ wv,   bf16_t* __restrict__ u_bf) {
  const int S0 = 512*512, S1 = 512*512, S2 = 96*512, S3 = 512*96, S4 = 96*512;
  const int total = S0 + S1 + S2 + S3 + S4 + 1024;
  for (int i0 = blockIdx.x * blockDim.x + threadIdx.x; i0 < total;
       i0 += gridDim.x * blockDim.x) {
    int i = i0;
    if (i < S0) { int r = i >> 9, c = i & 511;                 // wuuT[j][i] = wuu[i][j]
      wuuT[i] = (bf16_t)wuu[c * 512 + r]; continue; }
    i -= S0;
    if (i < S1) { int r = i >> 9, c = i & 511;                 // wuhT[n][k] = wuh[k][n]
      wuhT[i] = (bf16_t)wuh[c * 512 + r]; continue; }
    i -= S1;
    if (i < S2) { int r = i >> 9, c = i & 511;                 // wuvT[n][k] = wuv[k][n]
      wuvT[i] = (r < VD) ? (bf16_t)wuv[c * VD + r] : (bf16_t)0.f; continue; }
    i -= S2;
    if (i < S3) { int r = i / 96, c = i % 96;                  // wt[hd][vd] = w[vd][hd]
      wt[i] = (c < VD) ? (bf16_t)w[c * 512 + r] : (bf16_t)0.f; continue; }
    i -= S3;
    if (i < S4) { int r = i >> 9, c = i & 511;                 // wv[vd][hd] = w[vd][hd]
      wv[i] = (r < VD) ? (bf16_t)w[r * 512 + c] : (bf16_t)0.f; continue; }
    i -= S4;
    if (i < 512) u_bf[i] = (bf16_t)u0[i];                      // u row 0 = u0
    else u_bf[(size_t)16383 * 512 + (i - 512)] = (bf16_t)0.f;  // pad row 16383
  }
}

// ---------------------------------------------------------------------------
// cin[t][j] = bu[j] + visible[t] . wvu[:,j]   (t = 1..16382)
// ---------------------------------------------------------------------------
__global__ void cin_kernel(const float* __restrict__ visible, const float* __restrict__ wvu,
                           const float* __restrict__ bu, float* __restrict__ cin) {
  const int gid = blockIdx.x * blockDim.x + threadIdx.x;
  if (gid >= 16382 * 512) return;
  const int t = 1 + (gid >> 9);
  const int j = gid & 511;
  float acc = bu[j];
  const float* vrow = visible + (size_t)t * VD;
#pragma unroll 8
  for (int i = 0; i < VD; ++i) acc = fmaf(vrow[i], wvu[(size_t)i * HD + j], acc);
  cin[(size_t)t * HD + j] = acc;
}

// ---------------------------------------------------------------------------
// sequential RNN scan: u_t = tanh(cin[t] + u_{t-1} @ wuu)
// 1 block x 1024 threads; u lives in LDS; wuu cols [0,ISPLIT) pinned in LDS via
// a 2D TDM tensor load, remainder streamed from L2 as bf16.
// ---------------------------------------------------------------------------
__global__ void __launch_bounds__(1024)
scan_kernel(const float* __restrict__ cin, const bf16_t* __restrict__ wuuT,
            const float* __restrict__ u0, bf16_t* __restrict__ u_bf) {
  extern __shared__ char ssm[];
  bf16_t* wlds = (bf16_t*)ssm;                               // [512][ISPLIT]
  float*  u_s  = (float*)(ssm + (size_t)512 * ISPLIT * 2);   // [512]
  float*  part = u_s + 512;                                  // [1024]
  const int tid = threadIdx.x;
  const int j = tid & 511, half = tid >> 9;
#if HAVE_TDM
  if (tid < 32) {   // wave 0 issues the DMA: 512 rows x ISPLIT cols, stride 512
    tdm_load_2d(0u, wuuT, ISPLIT, 512u, 512u, 512u, 512u);
    __builtin_amdgcn_s_wait_tensorcnt(0);
  }
#else
  for (int idx = tid; idx < 512 * (ISPLIT / 8); idx += 1024) {
    int r = idx / (ISPLIT / 8), c = idx % (ISPLIT / 8);
    ((uint4*)(wlds + (size_t)r * ISPLIT))[c] =
        ((const uint4*)(wuuT + (size_t)r * 512))[c];
  }
#endif
  if (tid < 512) u_s[tid] = u0[tid];
  __syncthreads();
  const bf16_t* lrow = wlds + (size_t)j * ISPLIT;
  const bf16_t* grow = wuuT + (size_t)j * 512 + ISPLIT;
  for (int t = 1; t <= 16382; ++t) {
    float acc = 0.f;
    if (half == 0) {            // i in [0,256) from LDS
#pragma unroll 4
      for (int ii = 0; ii < 256; ii += 8) {
        v8bf w8 = *reinterpret_cast<const v8bf*>(lrow + ii);
#pragma unroll
        for (int e = 0; e < 8; ++e) acc = fmaf((float)w8[e], u_s[ii + e], acc);
      }
    } else {                    // i in [256,ISPLIT) from LDS, rest from global
#pragma unroll
      for (int ii = 256; ii < ISPLIT; ii += 8) {
        v8bf w8 = *reinterpret_cast<const v8bf*>(lrow + ii);
#pragma unroll
        for (int e = 0; e < 8; ++e) acc = fmaf((float)w8[e], u_s[ii + e], acc);
      }
#pragma unroll 4
      for (int ii = 0; ii < 512 - ISPLIT; ii += 8) {
        v8bf w8 = *reinterpret_cast<const v8bf*>(grow + ii);
#pragma unroll
        for (int e = 0; e < 8; ++e) acc = fmaf((float)w8[e], u_s[ISPLIT + ii + e], acc);
      }
    }
    part[tid] = acc;
    __syncthreads();
    if (tid < 512) {
      float un = tanhf(cin[(size_t)t * RD + j] + part[j] + part[j + 512]);
      u_s[j] = un;
      u_bf[(size_t)t * RD + j] = (bf16_t)un;
    }
    __syncthreads();
  }
}

// ---------------------------------------------------------------------------
// bh_t = u @ wuh + bh ; bv_t = u @ wuv + bv   (WMMA bf16, one wave per tile)
// ---------------------------------------------------------------------------
__global__ void __launch_bounds__(256)
biasgemm_kernel(const bf16_t* __restrict__ u_bf,
                const bf16_t* __restrict__ wuhT, const bf16_t* __restrict__ wuvT,
                const float* __restrict__ bh, const float* __restrict__ bv,
                float* __restrict__ bh_t, float* __restrict__ bv_t) {
  const int wid = threadIdx.x >> 5, lane = threadIdx.x & 31;
  const int idx = lane & 15, g = lane >> 4;
  const int wg = blockIdx.x * 8 + wid;
  if (wg < 1024 * 32) {                     // ---- bh_t: N = 512 ----
    const int mt = wg >> 5, nt = wg & 31;
    v8f acc;
#pragma unroll
    for (int v = 0; v < 8; ++v) acc[v] = bh[nt * 16 + idx];
    for (int kc = 0; kc < 16; ++kc) {
      v16bf a = load_fragA(u_bf, mt * 16, HD, kc * 32, lane);
      v16bf b = load_fragB(wuhT, nt * 16, HD, kc * 32, lane);
      acc = wmma_bf16(a, b, acc);
    }
#pragma unroll
    for (int v = 0; v < 8; ++v)
      bh_t[(size_t)(mt * 16 + v + 8 * g) * HD + nt * 16 + idx] = acc[v];
  } else {                                  // ---- bv_t: N = 96 (padded) ----
    const int wg2 = wg - 1024 * 32;
    if (wg2 >= 1024 * 6) return;
    const int mt = wg2 / 6, nt = wg2 % 6;
    v8f acc;
#pragma unroll
    for (int v = 0; v < 8; ++v) {
      int c = nt * 16 + idx;
      acc[v] = (c < VD) ? bv[c] : 0.f;
    }
    for (int kc = 0; kc < 16; ++kc) {
      v16bf a = load_fragA(u_bf, mt * 16, HD, kc * 32, lane);
      v16bf b = load_fragB(wuvT, nt * 16, HD, kc * 32, lane);
      acc = wmma_bf16(a, b, acc);
    }
#pragma unroll
    for (int v = 0; v < 8; ++v)
      bv_t[(size_t)(mt * 16 + v + 8 * g) * VDP + nt * 16 + idx] = acc[v];
  }
}

// ---------------------------------------------------------------------------
// cross-entropy cost -> d_out[0]
// ---------------------------------------------------------------------------
__global__ void cost_kernel(const float* __restrict__ visible,
                            const float* __restrict__ bv_t, float* __restrict__ d_out) {
  __shared__ float red[256];
  const int tid = threadIdx.x;
  float local = 0.f;
  const size_t N = (size_t)16383 * VD;
  for (size_t i = (size_t)blockIdx.x * blockDim.x + tid; i < N;
       i += (size_t)gridDim.x * blockDim.x) {
    int r = (int)(i / VD), c = (int)(i % VD);
    float y = sigmoidf(bv_t[(size_t)r * VDP + c]);
    float v = visible[(size_t)(r + 1) * VD + c];
    local += -v * logf(EPSV + y) - (1.f - v) * logf(EPSV + 1.f - y);
  }
  red[tid] = local;
  __syncthreads();
  for (int s = 128; s > 0; s >>= 1) {
    if (tid < s) red[tid] += red[tid + s];
    __syncthreads();
  }
  if (tid == 0) atomicAdd(d_out, red[0] * (1.0f / 16384.0f));
}

// ---------------------------------------------------------------------------
// 20-step Gibbs chain, fully row-parallel. One wave per 16-row tile.
// LDS: w (HD-major) 96KB + w (VD-major) 96KB shared (TDM-loaded);
// per-wave h(16KB)+v(3KB).
// ---------------------------------------------------------------------------
__global__ void __launch_bounds__(192)
gibbs_kernel(const float* __restrict__ visible,
             const float* __restrict__ bh_t, const float* __restrict__ bv_t,
             const bf16_t* __restrict__ wt_g, const bf16_t* __restrict__ wv_g,
             float* __restrict__ d_out) {
  extern __shared__ char smem[];
  bf16_t* wt  = (bf16_t*)smem;        // [512][96]  B for h = v@w
  bf16_t* wvm = wt + 512 * 96;        // [96][512]  B for v = h@w^T
  const int tid = threadIdx.x;
#if HAVE_TDM
  if (tid < 32) {   // wave 0 DMAs both weight images into LDS
    tdm_load_2d(0u,      wt_g, 49152u, 1u, 49152u, 1u, 49152u);
    tdm_load_2d(98304u,  wv_g, 49152u, 1u, 49152u, 1u, 49152u);
    __builtin_amdgcn_s_wait_tensorcnt(0);
  }
#else
  {
    const uint4* s0 = (const uint4*)wt_g;  uint4* d0 = (uint4*)wt;
    for (int i = tid; i < (512 * 96) / 8; i += 192) d0[i] = s0[i];
    const uint4* s1 = (const uint4*)wv_g;  uint4* d1 = (uint4*)wvm;
    for (int i = tid; i < (96 * 512) / 8; i += 192) d1[i] = s1[i];
  }
#endif
  __syncthreads();
  const int wid = tid >> 5, lane = tid & 31;
  const int mt = blockIdx.x * GW + wid;
  if (mt >= 1024) return;
  const int r0 = mt * 16;
  bf16_t* hbuf = (bf16_t*)(smem + 2 * 512 * 96 * 2) + (size_t)wid * (16 * 512 + 16 * 96);
  bf16_t* vbuf = hbuf + 16 * 512;
  for (int i = lane; i < 16 * 96; i += 32) {        // v init = visible[:-1]
    int m = i / 96, c = i - m * 96, r = r0 + m;
    float val = (c < VD && r < 16383) ? visible[(size_t)r * VD + c] : 0.f;
    vbuf[i] = (bf16_t)val;
  }
  const int idx = lane & 15, g = lane >> 4;
  float msum[8] = {0, 0, 0, 0, 0, 0, 0, 0};
  for (int s = 0; s < GIBBS; ++s) {
    // ---- h = sample(sigmoid(v @ w + bh_t)), K = 96 (3 chunks) ----
    v16bf a0 = load_fragA(vbuf, 0, VDP, 0, lane);
    v16bf a1 = load_fragA(vbuf, 0, VDP, 32, lane);
    v16bf a2 = load_fragA(vbuf, 0, VDP, 64, lane);
    for (int nt = 0; nt < 32; ++nt) {
      v8f c;
#pragma unroll
      for (int v = 0; v < 8; ++v)
        c[v] = bh_t[(size_t)(r0 + v + 8 * g) * HD + nt * 16 + idx];
      c = wmma_bf16(a0, load_fragB(wt, nt * 16, VDP, 0, lane), c);
      c = wmma_bf16(a1, load_fragB(wt, nt * 16, VDP, 32, lane), c);
      c = wmma_bf16(a2, load_fragB(wt, nt * 16, VDP, 64, lane), c);
#pragma unroll
      for (int v = 0; v < 8; ++v) {
        int row = r0 + v + 8 * g, col = nt * 16 + idx;
        float p = sigmoidf(c[v]);
        float samp = (p > urand(row, col, s, 0)) ? 1.f : 0.f;
        hbuf[(v + 8 * g) * HD + col] = (bf16_t)samp;
      }
    }
    // ---- v = sample(sigmoid(h @ w^T + bv_t)), K = 512 (16 chunks) ----
    v8f acc[6];
#pragma unroll
    for (int nt = 0; nt < 6; ++nt)
#pragma unroll
      for (int v = 0; v < 8; ++v)
        acc[nt][v] = bv_t[(size_t)(r0 + v + 8 * g) * VDP + nt * 16 + idx];
    for (int kc = 0; kc < 16; ++kc) {
      v16bf a = load_fragA(hbuf, 0, HD, kc * 32, lane);
#pragma unroll
      for (int nt = 0; nt < 6; ++nt)
        acc[nt] = wmma_bf16(a, load_fragB(wvm, nt * 16, HD, kc * 32, lane), acc[nt]);
    }
#pragma unroll
    for (int nt = 0; nt < 6; ++nt)
#pragma unroll
      for (int v = 0; v < 8; ++v) {
        int row = r0 + v + 8 * g, col = nt * 16 + idx;
        float p = sigmoidf(acc[nt][v]);
        float samp = (p > urand(row, col, s, 1)) ? 1.f : 0.f;
        vbuf[(v + 8 * g) * VDP + col] = (bf16_t)samp;
        if (s == GIBBS - 1 && col < VD && row < 16383)
          msum[v] += fabsf(visible[(size_t)(row + 1) * VD + col] - samp);
      }
  }
  // reduce across the 16 lanes of each half (rows m = v + 8g)
#pragma unroll
  for (int off = 1; off < 16; off <<= 1)
#pragma unroll
    for (int v = 0; v < 8; ++v) msum[v] += __shfl_xor(msum[v], off, 32);
  if (idx == 0) {
#pragma unroll
    for (int v = 0; v < 8; ++v) {
      int row = r0 + v + 8 * g;
      if (row < 16383) d_out[1 + row] = msum[v] * (1.0f / VD);
    }
  }
}

// ---------------------------------------------------------------------------
extern "C" void kernel_launch(void* const* d_in, const int* in_sizes, int n_in,
                              void* d_out, int out_size, void* d_ws, size_t ws_size,
                              hipStream_t stream) {
  const float* visible = (const float*)d_in[0];
  const float* w       = (const float*)d_in[1];
  const float* wuu     = (const float*)d_in[2];
  const float* wuv     = (const float*)d_in[3];
  const float* wuh     = (const float*)d_in[4];
  const float* wvu     = (const float*)d_in[5];
  const float* bv      = (const float*)d_in[6];
  const float* bh      = (const float*)d_in[7];
  const float* bu      = (const float*)d_in[8];
  const float* u0      = (const float*)d_in[9];
  float* out = (float*)d_out;
  char*  ws  = (char*)d_ws;

  float*  cin  = (float*)(ws + OFF_CIN);
  bf16_t* u_bf = (bf16_t*)(ws + OFF_UBF);
  float*  bh_t = (float*)(ws + OFF_BH);
  float*  bv_t = (float*)(ws + OFF_BV);
  bf16_t* wuuT = (bf16_t*)(ws + OFF_WUUT);
  bf16_t* wuhT = (bf16_t*)(ws + OFF_WUHT);
  bf16_t* wuvT = (bf16_t*)(ws + OFF_WUVT);
  bf16_t* wt   = (bf16_t*)(ws + OFF_WT);
  bf16_t* wv   = (bf16_t*)(ws + OFF_WV);

  hipMemsetAsync(d_out, 0, sizeof(float), stream);
  prep_kernel<<<1024, 256, 0, stream>>>(w, wuu, wuv, wuh, u0,
                                        wuuT, wuhT, wuvT, wt, wv, u_bf);
  cin_kernel<<<(16382 * 512) / 256, 256, 0, stream>>>(visible, wvu, bu, cin);

  const int scan_smem = 512 * ISPLIT * 2 + 512 * 4 + 1024 * 4;   // 317,440 B
  hipFuncSetAttribute(reinterpret_cast<const void*>(scan_kernel),
                      hipFuncAttributeMaxDynamicSharedMemorySize, scan_smem);
  scan_kernel<<<1, 1024, scan_smem, stream>>>(cin, wuuT, u0, u_bf);

  biasgemm_kernel<<<4864, 256, 0, stream>>>(u_bf, wuhT, wuvT, bh, bv, bh_t, bv_t);
  cost_kernel<<<1024, 256, 0, stream>>>(visible, bv_t, out);

  const int gibbs_smem = 2 * 512 * 96 * 2 + GW * (16 * 512 + 16 * 96) * 2; // 313,344 B
  hipFuncSetAttribute(reinterpret_cast<const void*>(gibbs_kernel),
                      hipFuncAttributeMaxDynamicSharedMemorySize, gibbs_smem);
  gibbs_kernel<<<171, 192, gibbs_smem, stream>>>(visible, bh_t, bv_t, wt, wv, out);

  (void)in_sizes; (void)n_in; (void)out_size; (void)ws_size;
}
